// BertLayer_4234837754336
// MI455X (gfx1250) — matrix operand order repaired and checked by reading, not compiled, not run on hardware
//
#include <hip/hip_runtime.h>

// ---------------- problem constants ----------------
constexpr int kH  = 768;
constexpr int kNH = 12;
constexpr int kDK = 64;
constexpr int kB  = 2;
constexpr int kT  = 2048;
constexpr int kP4 = 4096;      // pos rows padded 4095 -> 4096 (last row zero)
constexpr int kFF = 4 * kH;    // 3072
constexpr int kNTOK = kB * kT; // 4096
constexpr int kBH = kB * kNH;  // 24

typedef __attribute__((ext_vector_type(16))) __bf16 v16bf;
typedef __attribute__((ext_vector_type(8)))  float  v8f;
typedef __attribute__((address_space(3))) unsigned char lds_uchar;

union Frag { uint4 q[2]; v16bf b; };

__device__ __forceinline__ unsigned int rne_bf16_bits(float f) {
  unsigned int u = __builtin_bit_cast(unsigned int, f);
  return u + (0x7fffu + ((u >> 16) & 1u));   // round-to-nearest-even in [31:16]
}
__device__ __forceinline__ unsigned short f32_to_bf16(float f) {
  return (unsigned short)(rne_bf16_bits(f) >> 16);
}
// single v_perm_b32 packs the two rounded high-halves
__device__ __forceinline__ unsigned int pack_bf16x2(float lo, float hi) {
  return __builtin_amdgcn_perm(rne_bf16_bits(hi), rne_bf16_bits(lo), 0x07060302u);
}

__device__ __forceinline__ v8f wmma_bf16(const Frag& a, const Frag& b, v8f c) {
  // D(16x16,f32) = A(16x32,bf16) * B(32x16,bf16) + C
  return __builtin_amdgcn_wmma_f32_16x16x32_bf16(false, a.b, false, b.b,
                                                 (short)0, c, false, false);
}

__device__ __forceinline__ float gelu_exact(float x) {
  return 0.5f * x * (1.0f + erff(x * 0.70710678118654752f));
}

// reductions confined to 16-lane half-wave groups (matches WMMA C layout)
__device__ __forceinline__ float group_max16(float v) {
  #pragma unroll
  for (int m = 1; m <= 8; m <<= 1) v = fmaxf(v, __shfl_xor(v, m, 32));
  return v;
}
__device__ __forceinline__ float group_sum16(float v) {
  #pragma unroll
  for (int m = 1; m <= 8; m <<= 1) v += __shfl_xor(v, m, 32);
  return v;
}

// =====================================================================
// Generic GEMM: C[M,N] = act(A[M,K] @ W[K,N] + bias), fp32 in/out,
// bf16 WMMA compute with f32 accumulate. BM=128 BN=64 BK=32, 8 waves.
// FULL/ACT are template params: the common path carries no row guards.
// =====================================================================
template <bool FULL, int ACT>
__global__ void __launch_bounds__(256)
gemm_bias_act_t(const float* __restrict__ A, const float* __restrict__ W,
                const float* __restrict__ bias, float* __restrict__ C,
                int M, int N, int K) {
  constexpr int BM = 128, BN = 64, BK = 32, LDP = BK + 8; // LDS stride 40
  __shared__ __attribute__((aligned(16))) unsigned short As[BM][LDP];
  __shared__ __attribute__((aligned(16))) unsigned short Bs[BN][LDP]; // [n][k]

  const int tid  = threadIdx.x;
  const int lane = tid & 31, w = tid >> 5;
  const int wm = w & 3, wn = w >> 2;        // 4x2 wave grid of 32x32 tiles
  const int g = lane >> 4, ln = lane & 15;
  const int m0 = blockIdx.x * BM, n0 = blockIdx.y * BN;

  // B staging micro-tile: 4k x 2n per thread
  const int skr = (tid >> 5) << 2;          // k row base 0..28
  const int snc = (tid & 31) << 1;          // n col base 0..62

  v8f acc[2][2] = {};

  for (int k0 = 0; k0 < K; k0 += BK) {
    // ---- A tile (128x32): 4 x float4 per thread, b64 LDS stores ----
    #pragma unroll
    for (int i = 0; i < 4; ++i) {
      const int idx = tid + i * 256;        // 1024 float4s
      const int r = idx >> 3, c = (idx & 7) << 2;
      float4 v;
      if (FULL) {
        v = *(const float4*)(A + (size_t)(m0 + r) * K + k0 + c);
      } else {
        const int gr = m0 + r;
        const int cr = gr < M ? gr : (M - 1);
        v = *(const float4*)(A + (size_t)cr * K + k0 + c);
        if (gr >= M) { v.x = 0.f; v.y = 0.f; v.z = 0.f; v.w = 0.f; }
      }
      uint2 p;
      p.x = pack_bf16x2(v.x, v.y);
      p.y = pack_bf16x2(v.z, v.w);
      *(uint2*)&As[r][c] = p;
    }
    // ---- B tile (32x64) transposed into [n][k]: 4k x 2n micro-tile ----
    {
      const float* Wp0 = W + (size_t)(k0 + skr) * N + n0 + snc;
      const float2 w0 = *(const float2*)(Wp0);
      const float2 w1 = *(const float2*)(Wp0 + N);
      const float2 w2 = *(const float2*)(Wp0 + 2 * N);
      const float2 w3 = *(const float2*)(Wp0 + 3 * N);
      uint2 c0, c1;
      c0.x = pack_bf16x2(w0.x, w1.x); c0.y = pack_bf16x2(w2.x, w3.x);
      c1.x = pack_bf16x2(w0.y, w1.y); c1.y = pack_bf16x2(w2.y, w3.y);
      *(uint2*)&Bs[snc][skr]     = c0;
      *(uint2*)&Bs[snc + 1][skr] = c1;
    }
    // prefetch next K tile into caches (global_prefetch_b8)
    if (k0 + BK < K) {
      const int pr = m0 + (tid >> 1);
      if (FULL || pr < M)
        __builtin_prefetch(&A[(size_t)pr * K + k0 + BK + (tid & 1) * 16], 0, 1);
      __builtin_prefetch(&W[(size_t)(k0 + BK + (tid >> 3)) * N + n0 + (tid & 7) * 8], 0, 1);
    }
    __syncthreads();

    #pragma unroll
    for (int i = 0; i < 2; ++i) {
      Frag a;
      const int am = wm * 32 + i * 16 + ln;
      a.q[0] = *(const uint4*)&As[am][16 * g];      // K = 16g..16g+7
      a.q[1] = *(const uint4*)&As[am][16 * g + 8];  // K = 16g+8..16g+15
      #pragma unroll
      for (int j = 0; j < 2; ++j) {
        Frag bb;
        const int bn = wn * 32 + j * 16 + ln;
        bb.q[0] = *(const uint4*)&Bs[bn][16 * g];   // K = 16g..16g+15
        bb.q[1] = *(const uint4*)&Bs[bn][16 * g + 8];
        acc[i][j] = wmma_bf16(a, bb, acc[i][j]);
      }
    }
    __syncthreads();
  }

  #pragma unroll
  for (int i = 0; i < 2; ++i)
    #pragma unroll
    for (int j = 0; j < 2; ++j)
      #pragma unroll
      for (int v = 0; v < 8; ++v) {
        const int m = m0 + wm * 32 + i * 16 + v + 8 * g;
        const int n = n0 + wn * 32 + j * 16 + ln;
        if (FULL || m < M) {
          float x = acc[i][j][v] + (bias ? bias[n] : 0.f);
          if (ACT == 1) x = gelu_exact(x);
          C[(size_t)m * N + n] = x;
        }
      }
}

// =====================================================================
// Pack kernels (vectorized x4): build bf16 WMMA-friendly layouts.
// =====================================================================
__global__ void __launch_bounds__(256)
pack_qkv(const float* __restrict__ qf, const float* __restrict__ kf,
         const float* __restrict__ vf, const float* __restrict__ pbu,
         const float* __restrict__ pbv, unsigned short* __restrict__ quh,
         unsigned short* __restrict__ qvh, unsigned short* __restrict__ khh,
         unsigned short* __restrict__ vth) {
  const int idx = blockIdx.x * blockDim.x + threadIdx.x;  // over NTOK*H/4
  if (idx >= kNTOK * kH / 4) return;
  const int e  = idx * 4;
  const int h_ = e % kH;
  const int t  = (e / kH) % kT;
  const int b  = e / (kH * kT);
  const int d  = h_ % kDK, hh = h_ / kDK;

  const float4 q4 = *(const float4*)(qf + e);
  const float4 k4 = *(const float4*)(kf + e);
  const float4 v4 = *(const float4*)(vf + e);
  const float4 u4 = *(const float4*)(pbu + hh * kDK + d);
  const float4 w4 = *(const float4*)(pbv + hh * kDK + d);

  const size_t bh  = (size_t)(b * kNH + hh);
  const size_t dst = (bh * kT + t) * kDK + d;
  uint2 p;
  p.x = pack_bf16x2(q4.x + u4.x, q4.y + u4.y);
  p.y = pack_bf16x2(q4.z + u4.z, q4.w + u4.w);
  *(uint2*)(quh + dst) = p;
  p.x = pack_bf16x2(q4.x + w4.x, q4.y + w4.y);
  p.y = pack_bf16x2(q4.z + w4.z, q4.w + w4.w);
  *(uint2*)(qvh + dst) = p;
  p.x = pack_bf16x2(k4.x, k4.y);
  p.y = pack_bf16x2(k4.z, k4.w);
  *(uint2*)(khh + dst) = p;
  // V transposed [BH][DK][T] (scattered b16 stores, stride T)
  const size_t vt = (bh * kDK + d) * kT + t;
  vth[vt]          = f32_to_bf16(v4.x);
  vth[vt + kT]     = f32_to_bf16(v4.y);
  vth[vt + 2 * kT] = f32_to_bf16(v4.z);
  vth[vt + 3 * kT] = f32_to_bf16(v4.w);
}

__global__ void __launch_bounds__(256)
pack_p(const float* __restrict__ pf, unsigned short* __restrict__ ph) {
  const int idx = blockIdx.x * blockDim.x + threadIdx.x;    // [NH][4096][DK]/4
  if (idx >= kNH * kP4 * kDK / 4) return;
  const int e  = idx * 4;
  const int d  = e % kDK;
  const int r  = (e / kDK) % kP4;
  const int hh = e / (kDK * kP4);
  const int rc = r < (2 * kT - 1) ? r : (2 * kT - 2);       // branchless clamp
  float4 v = *(const float4*)(pf + (size_t)rc * kH + hh * kDK + d);
  if (r >= 2 * kT - 1) { v.x = 0.f; v.y = 0.f; v.z = 0.f; v.w = 0.f; }
  uint2 p;
  p.x = pack_bf16x2(v.x, v.y);
  p.y = pack_bf16x2(v.z, v.w);
  *(uint2*)(ph + e) = p;
}

// =====================================================================
// Fused flash attention with Transformer-XL rel-shift.
// grid: (T/64, B*NH), block: 128 (4 waves, one 16-row q-tile each).
// scores tile 16x32 per step; BD via 16x48 band GEMM + LDS diagonal read.
// Mask tile copied global->LDS with the CDNA5 async-LDS path (ASYNCcnt),
// overlapping the WMMA work.
// =====================================================================
__global__ void __launch_bounds__(128)
attn_kernel(const unsigned short* __restrict__ quh,
            const unsigned short* __restrict__ qvh,
            const unsigned short* __restrict__ khh,
            const unsigned short* __restrict__ vth,
            const unsigned short* __restrict__ ph,
            const unsigned char* __restrict__ mask,
            float* __restrict__ ctx) {
  __shared__ __attribute__((aligned(16))) float          bd_lds[4][16][48];
  __shared__ __attribute__((aligned(16))) unsigned short pr_lds[4][16][32];
  __shared__ __attribute__((aligned(16))) unsigned char  mask_lds[4][16][32];

  const int lane = threadIdx.x & 31;
  const int w    = threadIdx.x >> 5;
  const int g    = lane >> 4, ln = lane & 15;
  const int bh   = blockIdx.y;
  const int b = bh / kNH, head = bh % kNH;
  const int q0 = blockIdx.x * 64 + w * 16;

  const unsigned short* Qu = quh + (size_t)bh * kT * kDK;
  const unsigned short* Qv = qvh + (size_t)bh * kT * kDK;
  const unsigned short* Kh = khh + (size_t)bh * kT * kDK;
  const unsigned short* Vt = vth + (size_t)bh * kDK * kT;
  const unsigned short* Ph = ph  + (size_t)head * kP4 * kDK;

  // LDS byte offset of this lane's 16B mask slot (for async-LDS copy)
  const unsigned mlofs = (unsigned)(unsigned long long)
      (lds_uchar*)&mask_lds[w][ln][16 * g];
  const unsigned char* mrow = mask + ((size_t)b * kT + (q0 + ln)) * kT + 16 * g;

  // Q fragments (A operands), resident for the whole k-loop
  Frag aqu[2], aqv[2];
  {
    const size_t qoff = (size_t)(q0 + ln) * kDK;
    #pragma unroll
    for (int kb = 0; kb < 2; ++kb) {
      const int d0 = kb * 32 + 16 * g;
      aqu[kb].q[0] = *(const uint4*)(Qu + qoff + d0);
      aqu[kb].q[1] = *(const uint4*)(Qu + qoff + d0 + 8);
      aqv[kb].q[0] = *(const uint4*)(Qv + qoff + d0);
      aqv[kb].q[1] = *(const uint4*)(Qv + qoff + d0 + 8);
    }
  }

  v8f oacc[4] = {};
  float rmax[8], rsum[8];
  #pragma unroll
  for (int v = 0; v < 8; ++v) { rmax[v] = -3.0e38f; rsum[v] = 0.f; }

  for (int j0 = 0; j0 < kT; j0 += 32) {
    // ---- async copy of this wave's 16x32 mask tile into LDS ----
    {
      // prior-tile LDS reads must have completed before overwriting
      asm volatile("s_wait_dscnt 0x0" ::: "memory");
      asm volatile("global_load_async_to_lds_b128 %0, %1, off"
                   :: "v"(mlofs), "v"(mrow + j0) : "memory");
    }
    // ---- AC = Qu @ K^T : 16x32 tile ----
    v8f ac[2] = {};
    #pragma unroll
    for (int ns = 0; ns < 2; ++ns) {
      const size_t joff = (size_t)(j0 + ns * 16 + ln) * kDK;
      #pragma unroll
      for (int kb = 0; kb < 2; ++kb) {
        Frag bb;
        const int d0 = kb * 32 + 16 * g;
        bb.q[0] = *(const uint4*)(Kh + joff + d0);
        bb.q[1] = *(const uint4*)(Kh + joff + d0 + 8);
        ac[ns] = wmma_bf16(aqu[kb], bb, ac[ns]);
      }
    }
    // ---- BD band: tmp[q][u] = Qv[q] . P[r0+u], u in [0,48) ----
    const int r0 = 2032 - q0 + j0;   // = (T-1) - (q0+15) + j0
    #pragma unroll
    for (int ns = 0; ns < 3; ++ns) {
      v8f bd = {};
      const size_t roff = (size_t)(r0 + ns * 16 + ln) * kDK;
      #pragma unroll
      for (int kb = 0; kb < 2; ++kb) {
        Frag bb;
        const int d0 = kb * 32 + 16 * g;
        bb.q[0] = *(const uint4*)(Ph + roff + d0);
        bb.q[1] = *(const uint4*)(Ph + roff + d0 + 8);
        bd = wmma_bf16(aqv[kb], bb, bd);
      }
      #pragma unroll
      for (int v = 0; v < 8; ++v) bd_lds[w][v + 8 * g][ns * 16 + ln] = bd[v];
    }
    // mask tile must be in LDS before the combine
    asm volatile("s_wait_asynccnt 0x0" ::: "memory");
    // ---- combine (diagonal rel-shift read), scale, mask ----
    float s[2][8];
    #pragma unroll
    for (int ns = 0; ns < 2; ++ns)
      #pragma unroll
      for (int v = 0; v < 8; ++v) {
        const int m = v + 8 * g;
        const int jj = ns * 16 + ln;
        float sc = (ac[ns][v] + bd_lds[w][m][15 - m + jj]) * 0.125f; // 1/sqrt(64)
        if (mask_lds[w][m][jj]) sc = -10000.f;
        s[ns][v] = sc;
      }
    // ---- online softmax + stage probs as bf16 A-operand ----
    #pragma unroll
    for (int v = 0; v < 8; ++v) {
      float tm = group_max16(fmaxf(s[0][v], s[1][v]));
      const float nm   = fmaxf(rmax[v], tm);
      const float corr = __expf(rmax[v] - nm);
      rmax[v] = nm;
      const float p0 = __expf(s[0][v] - nm);
      const float p1 = __expf(s[1][v] - nm);
      rsum[v] = rsum[v] * corr + group_sum16(p0 + p1);
      #pragma unroll
      for (int ds = 0; ds < 4; ++ds) oacc[ds][v] *= corr;
      const int m = v + 8 * g;
      pr_lds[w][m][ln]      = f32_to_bf16(p0);
      pr_lds[w][m][16 + ln] = f32_to_bf16(p1);
    }
    // ---- O += P @ V (V^T layout gives contiguous key dim) ----
    Frag pa;
    pa.q[0] = *(const uint4*)&pr_lds[w][ln][16 * g];
    pa.q[1] = *(const uint4*)&pr_lds[w][ln][16 * g + 8];
    #pragma unroll
    for (int ds = 0; ds < 4; ++ds) {
      Frag bb;
      const size_t doff = (size_t)(ds * 16 + ln) * kT + j0 + 16 * g;
      bb.q[0] = *(const uint4*)(Vt + doff);
      bb.q[1] = *(const uint4*)(Vt + doff + 8);
      oacc[ds] = wmma_bf16(pa, bb, oacc[ds]);
    }
  }
  // ---- normalize and scatter into ctx[B,T,H] ----
  #pragma unroll
  for (int v = 0; v < 8; ++v) {
    const int m = v + 8 * g;
    const float inv = 1.f / rsum[v];
    #pragma unroll
    for (int ds = 0; ds < 4; ++ds)
      ctx[((size_t)b * kT + (q0 + m)) * kH + head * kDK + ds * 16 + ln] =
          oacc[ds][v] * inv;
  }
}

// =====================================================================
// out = LayerNorm(y + res) * g + b ; one block (192 thr, float4) per row
// =====================================================================
__global__ void __launch_bounds__(192)
add_layernorm(const float* __restrict__ y, const float* __restrict__ res,
              const float* __restrict__ gw, const float* __restrict__ bw,
              float* __restrict__ out) {
  const int row = blockIdx.x;
  const size_t base = (size_t)row * kH;
  const int t = threadIdx.x;
  const int c = t * 4;
  const float4 a  = *(const float4*)(y + base + c);
  const float4 r4 = *(const float4*)(res + base + c);
  float x0 = a.x + r4.x, x1 = a.y + r4.y, x2 = a.z + r4.z, x3 = a.w + r4.w;
  float s  = x0 + x1 + x2 + x3;
  float s2 = x0 * x0 + x1 * x1 + x2 * x2 + x3 * x3;
  #pragma unroll
  for (int m = 1; m < 32; m <<= 1) {
    s += __shfl_xor(s, m, 32);
    s2 += __shfl_xor(s2, m, 32);
  }
  __shared__ float rs[6], rs2[6];
  const int w = t >> 5, lane = t & 31;
  if (lane == 0) { rs[w] = s; rs2[w] = s2; }
  __syncthreads();
  s = 0.f; s2 = 0.f;
  #pragma unroll
  for (int i = 0; i < 6; ++i) { s += rs[i]; s2 += rs2[i]; }
  const float mu   = s * (1.f / kH);
  const float var  = s2 * (1.f / kH) - mu * mu;
  const float rstd = rsqrtf(var + 1e-5f);
  const float4 g4 = *(const float4*)(gw + c);
  const float4 b4 = *(const float4*)(bw + c);
  float4 o;
  o.x = (x0 - mu) * rstd * g4.x + b4.x;
  o.y = (x1 - mu) * rstd * g4.y + b4.y;
  o.z = (x2 - mu) * rstd * g4.z + b4.z;
  o.w = (x3 - mu) * rstd * g4.w + b4.w;
  *(float4*)(out + base + c) = o;
}

// =====================================================================
extern "C" void kernel_launch(void* const* d_in, const int* in_sizes, int n_in,
                              void* d_out, int out_size, void* d_ws, size_t ws_size,
                              hipStream_t stream) {
  (void)in_sizes; (void)n_in; (void)out_size; (void)ws_size;
  const float* x    = (const float*)d_in[0];
  const unsigned char* mask = (const unsigned char*)d_in[1];
  const float* pos  = (const float*)d_in[2];
  const float* Wq   = (const float*)d_in[3];
  const float* bq   = (const float*)d_in[4];
  const float* Wk   = (const float*)d_in[5];
  const float* bk   = (const float*)d_in[6];
  const float* Wv   = (const float*)d_in[7];
  const float* bv   = (const float*)d_in[8];
  const float* Wp   = (const float*)d_in[9];
  const float* pbu  = (const float*)d_in[10];
  const float* pbv  = (const float*)d_in[11];
  const float* Wo   = (const float*)d_in[12];
  const float* bo   = (const float*)d_in[13];
  const float* ln1g = (const float*)d_in[14];
  const float* ln1b = (const float*)d_in[15];
  const float* W1   = (const float*)d_in[16];
  const float* b1   = (const float*)d_in[17];
  const float* W2   = (const float*)d_in[18];
  const float* b2   = (const float*)d_in[19];
  const float* ln2g = (const float*)d_in[20];
  const float* ln2b = (const float*)d_in[21];
  float* out = (float*)d_out;

  char* ws = (char*)d_ws;
  size_t off = 0;
  auto alloc = [&](size_t bytes) -> void* {
    void* p = ws + off;
    off = (off + bytes + 255) & ~(size_t)255;
    return p;
  };
  float* qf = (float*)alloc((size_t)kNTOK * kH * 4);
  float* kf = (float*)alloc((size_t)kNTOK * kH * 4);
  float* vf = (float*)alloc((size_t)kNTOK * kH * 4);
  float* pf = (float*)alloc((size_t)(2 * kT - 1) * kH * 4);
  unsigned short* quh = (unsigned short*)alloc((size_t)kBH * kT * kDK * 2);
  unsigned short* qvh = (unsigned short*)alloc((size_t)kBH * kT * kDK * 2);
  unsigned short* khh = (unsigned short*)alloc((size_t)kBH * kT * kDK * 2);
  unsigned short* vth = (unsigned short*)alloc((size_t)kBH * kDK * kT * 2);
  unsigned short* phh = (unsigned short*)alloc((size_t)kNH * kP4 * kDK * 2);
  float* hbuf = (float*)alloc((size_t)kNTOK * kFF * 4);
  // buffer reuse (lifetimes are disjoint):
  float* ctx      = qf;  // q raw free after pack_qkv
  float* attn_out = kf;  // k raw free after pack_qkv
  float* x1       = vf;  // v raw free after pack_qkv
  float* y2       = qf;  // ctx free after Wo GEMM

  const dim3 g768((kNTOK + 127) / 128, kH / 64);
  const dim3 gpos((2 * kT - 1 + 127) / 128, kH / 64);
  const dim3 gff((kNTOK + 127) / 128, kFF / 64);

  // projections (M=4096 -> FULL path, no guards)
  gemm_bias_act_t<true, 0><<<g768, 256, 0, stream>>>(x, Wq, bq, qf, kNTOK, kH, kH);
  gemm_bias_act_t<true, 0><<<g768, 256, 0, stream>>>(x, Wk, bk, kf, kNTOK, kH, kH);
  gemm_bias_act_t<true, 0><<<g768, 256, 0, stream>>>(x, Wv, bv, vf, kNTOK, kH, kH);
  gemm_bias_act_t<false, 0><<<gpos, 256, 0, stream>>>(pos, Wp, nullptr, pf,
                                                      2 * kT - 1, kH, kH);
  // repack to bf16 WMMA layouts
  pack_qkv<<<(kNTOK * kH / 4 + 255) / 256, 256, 0, stream>>>(qf, kf, vf, pbu, pbv,
                                                             quh, qvh, khh, vth);
  pack_p<<<(kNH * kP4 * kDK / 4 + 255) / 256, 256, 0, stream>>>(pf, phh);
  // fused rel-pos flash attention
  attn_kernel<<<dim3(kT / 64, kBH), 128, 0, stream>>>(quh, qvh, khh, vth, phh,
                                                      mask, ctx);
  // output proj + residual LN
  gemm_bias_act_t<true, 0><<<g768, 256, 0, stream>>>(ctx, Wo, bo, attn_out,
                                                     kNTOK, kH, kH);
  add_layernorm<<<kNTOK, 192, 0, stream>>>(attn_out, x, ln1g, ln1b, x1);
  // FFN
  gemm_bias_act_t<true, 1><<<gff, 256, 0, stream>>>(x1, W1, b1, hbuf,
                                                    kNTOK, kFF, kH);
  gemm_bias_act_t<true, 0><<<g768, 256, 0, stream>>>(hbuf, W2, b2, y2,
                                                     kNTOK, kH, kFF);
  add_layernorm<<<kNTOK, 192, 0, stream>>>(y2, x1, ln2g, ln2b, out);
}
